// KSConditionalCrossAttentionDynamicAttn_91336774517600
// MI455X (gfx1250) — matrix operand order repaired and checked by reading, not compiled, not run on hardware
//
#include <hip/hip_runtime.h>

// Conditional-DETR cross attention (first-layer path) for MI455X / gfx1250.
// All matmuls use v_wmma_f32_16x16x32_f16 (wave32 WMMA), attention is fused
// flash-style (no materialized [B,H,900,4096] matrix).
// Weights are pre-converted to f16 once; activations are converted in-register.

#define NQ 900
#define NK 4096
#define BB 4
#define HH 8
#define CC 256
#define DD 32

typedef _Float16 h16 __attribute__((ext_vector_type(16)));
typedef _Float16 h8  __attribute__((ext_vector_type(8)));
typedef float    f8  __attribute__((ext_vector_type(8)));

#define WMMA_F16(a, b, c) \
  __builtin_amdgcn_wmma_f32_16x16x32_f16(false, (a), false, (b), (short)0, (c), false, false)

// ---- fragment loaders -------------------------------------------------------
// A fragment (16x32 f16): lane l -> row l%16, 16 consecutive K starting at k0+16g
static __device__ __forceinline__ h16 ldA_f32(const float* row, int k0, int g) {
  f8 c0 = *(const f8*)(row + k0 + 16 * g);
  f8 c1 = *(const f8*)(row + k0 + 16 * g + 8);
  h16 r;
#pragma unroll
  for (int i = 0; i < 8; ++i) { r[i] = (_Float16)c0[i]; r[i + 8] = (_Float16)c1[i]; }
  return r;
}
static __device__ __forceinline__ h16 ldA_f16(const _Float16* row, int k0, int g) {
  h8 c0 = *(const h8*)(row + k0 + 16 * g);
  h8 c1 = *(const h8*)(row + k0 + 16 * g + 8);
  return __builtin_shufflevector(c0, c1, 0, 1, 2, 3, 4, 5, 6, 7, 8, 9, 10, 11, 12, 13, 14, 15);
}
// B fragment (32x16 f16): lane l -> column l%16 = row of [NxK] storage,
// 16 consecutive K starting at k0 + 16*(l/16)
static __device__ __forceinline__ h16 ldB_f16(const _Float16* row, int k0, int g) {
  return *(const h16*)(row + k0 + 16 * g);
}

// ---- weight conversion f32 -> f16 ------------------------------------------
__global__ void cvt_f16_kernel(const float* __restrict__ src,
                               _Float16* __restrict__ dst, int n) {
  const int i = blockIdx.x * 256 + threadIdx.x;
  if (i < n) dst[i] = (_Float16)src[i];
}

// ---- kernel 1: q = query@Wqc^T + bqc + query_pos@Wqp^T + bqp -> qh[..][0:32]
//      (softmax scale 1/8 pre-folded into qh)
__global__ __launch_bounds__(32) void proj_q_kernel(
    const float* __restrict__ query, const float* __restrict__ qpos,
    const _Float16* __restrict__ Wqc, const _Float16* __restrict__ Wqp,
    const float* __restrict__ bqc, const float* __restrict__ bqp,
    _Float16* __restrict__ qh) {
  const int lane = threadIdx.x, g = lane >> 4, ln = lane & 15;
  const int ct4 = blockIdx.x & 3;            // 64-channel strip
  const int b   = (blockIdx.x >> 2) & 3;
  const int mt  = blockIdx.x >> 4;           // 0..56
  const int pos0 = mt * 16;
  const int posA = (pos0 + ln < NQ) ? (pos0 + ln) : (NQ - 1);
  const float* arow_q  = query + ((size_t)posA * BB + b) * CC;
  const float* arow_qp = qpos  + ((size_t)posA * BB + b) * CC;
  const int c0 = ct4 * 64;
  const _Float16* browc[4];
  const _Float16* browp[4];
#pragma unroll
  for (int j = 0; j < 4; ++j) {
    browc[j] = Wqc + (size_t)(c0 + 16 * j + ln) * CC;
    browp[j] = Wqp + (size_t)(c0 + 16 * j + ln) * CC;
  }
  f8 acc[4] = {};
#pragma unroll
  for (int k0 = 0; k0 < CC; k0 += 32) {
    const h16 aq = ldA_f32(arow_q,  k0, g);
    const h16 ap = ldA_f32(arow_qp, k0, g);
#pragma unroll
    for (int j = 0; j < 4; ++j) {
      acc[j] = WMMA_F16(aq, ldB_f16(browc[j], k0, g), acc[j]);
      acc[j] = WMMA_F16(ap, ldB_f16(browp[j], k0, g), acc[j]);
    }
  }
#pragma unroll
  for (int j = 0; j < 4; ++j) {
    const int c = c0 + 16 * j + ln;
    const float bias = bqc[c] + bqp[c];
    const int h = c >> 5, dd = c & 31;
    _Float16* outb = qh + ((size_t)(b * HH + h) * NQ) * 64 + dd;
#pragma unroll
    for (int v = 0; v < 8; ++v) {
      const int pos = pos0 + 8 * g + v;
      if (pos < NQ) outb[(size_t)pos * 64] = (_Float16)((acc[j][v] + bias) * 0.125f);
    }
  }
}

// ---- kernel 2: qse = query_sine_embed@Wqs^T + bqs -> qh[..][32:64] (scaled)
__global__ __launch_bounds__(32) void proj_qs_kernel(
    const float* __restrict__ qsine, const _Float16* __restrict__ Wqs,
    const float* __restrict__ bqs, _Float16* __restrict__ qh) {
  const int lane = threadIdx.x, g = lane >> 4, ln = lane & 15;
  const int ct4 = blockIdx.x & 3;
  const int b   = (blockIdx.x >> 2) & 3;
  const int mt  = blockIdx.x >> 4;
  const int pos0 = mt * 16;
  const int posA = (pos0 + ln < NQ) ? (pos0 + ln) : (NQ - 1);
  const float* arow = qsine + ((size_t)posA * BB + b) * CC;
  const int c0 = ct4 * 64;
  const _Float16* brow[4];
#pragma unroll
  for (int j = 0; j < 4; ++j) brow[j] = Wqs + (size_t)(c0 + 16 * j + ln) * CC;
  f8 acc[4] = {};
#pragma unroll
  for (int k0 = 0; k0 < CC; k0 += 32) {
    const h16 a = ldA_f32(arow, k0, g);
#pragma unroll
    for (int j = 0; j < 4; ++j)
      acc[j] = WMMA_F16(a, ldB_f16(brow[j], k0, g), acc[j]);
  }
#pragma unroll
  for (int j = 0; j < 4; ++j) {
    const int c = c0 + 16 * j + ln;
    const float bias = bqs[c];
    const int h = c >> 5, dd = c & 31;
    _Float16* outb = qh + ((size_t)(b * HH + h) * NQ) * 64 + 32 + dd;
#pragma unroll
    for (int v = 0; v < 8; ++v) {
      const int pos = pos0 + 8 * g + v;
      if (pos < NQ) outb[(size_t)pos * 64] = (_Float16)((acc[j][v] + bias) * 0.125f);
    }
  }
}

// ---- kernel 3: kp = key_pos@Wkp^T + bkp ; k = key@Wkc^T + bkc + kp
//      kh[..][0:32] = k ; kh[..][32:64] = kp
__global__ __launch_bounds__(32) void proj_k_kernel(
    const float* __restrict__ key, const float* __restrict__ kpos,
    const _Float16* __restrict__ Wkc, const _Float16* __restrict__ Wkp,
    const float* __restrict__ bkc, const float* __restrict__ bkp,
    _Float16* __restrict__ kh) {
  const int lane = threadIdx.x, g = lane >> 4, ln = lane & 15;
  const int ct4 = blockIdx.x & 3;
  const int b   = (blockIdx.x >> 2) & 3;
  const int mt  = blockIdx.x >> 4;           // 0..255
  const int pos0 = mt * 16;
  const float* arow_k  = key  + ((size_t)(pos0 + ln) * BB + b) * CC;
  const float* arow_kp = kpos + ((size_t)(pos0 + ln) * BB + b) * CC;
  const int c0 = ct4 * 64;
  const _Float16* browc[4];
  const _Float16* browp[4];
#pragma unroll
  for (int j = 0; j < 4; ++j) {
    browc[j] = Wkc + (size_t)(c0 + 16 * j + ln) * CC;
    browp[j] = Wkp + (size_t)(c0 + 16 * j + ln) * CC;
  }
  f8 acc_k[4] = {}, acc_p[4] = {};
#pragma unroll
  for (int k0 = 0; k0 < CC; k0 += 32) {
    const h16 aK = ldA_f32(arow_k,  k0, g);
    const h16 aP = ldA_f32(arow_kp, k0, g);
#pragma unroll
    for (int j = 0; j < 4; ++j) {
      const h16 bP = ldB_f16(browp[j], k0, g);
      acc_k[j] = WMMA_F16(aK, ldB_f16(browc[j], k0, g), acc_k[j]);
      acc_k[j] = WMMA_F16(aP, bP, acc_k[j]);
      acc_p[j] = WMMA_F16(aP, bP, acc_p[j]);
    }
  }
#pragma unroll
  for (int j = 0; j < 4; ++j) {
    const int c = c0 + 16 * j + ln;
    const float bp = bkp[c];
    const float bk = bkc[c] + bp;
    const int h = c >> 5, dd = c & 31;
    _Float16* outb = kh + ((size_t)(b * HH + h) * NK) * 64;
#pragma unroll
    for (int v = 0; v < 8; ++v) {
      const int pos = pos0 + 8 * g + v;
      outb[(size_t)pos * 64 + dd]      = (_Float16)(acc_k[j][v] + bk);
      outb[(size_t)pos * 64 + 32 + dd] = (_Float16)(acc_p[j][v] + bp);
    }
  }
}

// ---- kernel 4: v = value@Wv^T + bv -> vT[b][h][dim][key]  (transposed store)
__global__ __launch_bounds__(32) void proj_v_kernel(
    const float* __restrict__ value, const _Float16* __restrict__ Wv,
    const float* __restrict__ bv, _Float16* __restrict__ vT) {
  const int lane = threadIdx.x, g = lane >> 4, ln = lane & 15;
  const int ct4 = blockIdx.x & 3;
  const int b   = (blockIdx.x >> 2) & 3;
  const int mt  = blockIdx.x >> 4;           // 0..255
  const int pos0 = mt * 16;
  const float* arow = value + ((size_t)(pos0 + ln) * BB + b) * CC;
  const int c0 = ct4 * 64;
  const _Float16* brow[4];
#pragma unroll
  for (int j = 0; j < 4; ++j) brow[j] = Wv + (size_t)(c0 + 16 * j + ln) * CC;
  f8 acc[4] = {};
#pragma unroll
  for (int k0 = 0; k0 < CC; k0 += 32) {
    const h16 a = ldA_f32(arow, k0, g);
#pragma unroll
    for (int j = 0; j < 4; ++j)
      acc[j] = WMMA_F16(a, ldB_f16(brow[j], k0, g), acc[j]);
  }
#pragma unroll
  for (int j = 0; j < 4; ++j) {
    const int c = c0 + 16 * j + ln;
    const float bias = bv[c];
    const int h = c >> 5, dd = c & 31;
    _Float16* outb = vT + ((size_t)(b * HH + h) * DD + dd) * NK;
#pragma unroll
    for (int v = 0; v < 8; ++v)
      outb[pos0 + 8 * g + v] = (_Float16)(acc[j][v] + bias);
  }
}

// ---- kernel 5: fused flash attention per (b,h,16-query tile), 64 keys/step
__global__ __launch_bounds__(32) void attn_kernel(
    const _Float16* __restrict__ qh, const _Float16* __restrict__ kh,
    const _Float16* __restrict__ vT, _Float16* __restrict__ att) {
  __shared__ _Float16 pP[16 * 64];
  const int lane = threadIdx.x, g = lane >> 4, ln = lane & 15;
  const int qt = blockIdx.x % 57;
  const int h  = (blockIdx.x / 57) & 7;
  const int b  = blockIdx.x / (57 * 8);
  const int pos0 = qt * 16;
  const int posA = (pos0 + ln < NQ) ? (pos0 + ln) : (NQ - 1);

  const _Float16* qrow = qh + ((size_t)(b * HH + h) * NQ + posA) * 64;
  const h16 qa0 = ldA_f16(qrow, 0, g);
  const h16 qa1 = ldA_f16(qrow, 32, g);
  const _Float16* khbase = kh + ((size_t)(b * HH + h) * NK) * 64;
  const _Float16* vbase  = vT + ((size_t)(b * HH + h) * DD) * NK;

  f8 acc0 = {}, acc1 = {};
  float run_m[8], run_l[8];
#pragma unroll
  for (int v = 0; v < 8; ++v) { run_m[v] = -3.0e38f; run_l[v] = 0.0f; }

  for (int kb = 0; kb < NK; kb += 64) {
    // scores: four 16-key tiles, head-dim 64 = 2 WMMA K-steps each (scale in qh)
    f8 s[4];
#pragma unroll
    for (int i = 0; i < 4; ++i) {
      const _Float16* krow = khbase + (size_t)(kb + 16 * i + ln) * 64;
      f8 si = {};
      si = WMMA_F16(qa0, ldB_f16(krow, 0, g),  si);
      si = WMMA_F16(qa1, ldB_f16(krow, 32, g), si);
      s[i] = si;
    }

    // online softmax over 64 keys; row stats reduced across each 16-lane group
#pragma unroll
    for (int v = 0; v < 8; ++v) {
      const float a0 = s[0][v], a1 = s[1][v], a2 = s[2][v], a3 = s[3][v];
      float mx = fmaxf(fmaxf(a0, a1), fmaxf(a2, a3));
      mx = fmaxf(mx, __shfl_xor(mx, 1));
      mx = fmaxf(mx, __shfl_xor(mx, 2));
      mx = fmaxf(mx, __shfl_xor(mx, 4));
      mx = fmaxf(mx, __shfl_xor(mx, 8));
      const float nm   = fmaxf(run_m[v], mx);
      const float corr = __expf(run_m[v] - nm);
      const float p0 = __expf(a0 - nm), p1 = __expf(a1 - nm);
      const float p2 = __expf(a2 - nm), p3 = __expf(a3 - nm);
      float rs = (p0 + p1) + (p2 + p3);
      rs += __shfl_xor(rs, 1);
      rs += __shfl_xor(rs, 2);
      rs += __shfl_xor(rs, 4);
      rs += __shfl_xor(rs, 8);
      run_l[v] = run_l[v] * corr + rs;
      run_m[v] = nm;
      acc0[v] *= corr;
      acc1[v] *= corr;
      const int m = 8 * g + v;                 // C-layout row
      pP[m * 64 + ln]      = (_Float16)p0;
      pP[m * 64 + 16 + ln] = (_Float16)p1;
      pP[m * 64 + 32 + ln] = (_Float16)p2;
      pP[m * 64 + 48 + ln] = (_Float16)p3;
    }
    __syncthreads();
    // re-read P as two A fragments (16 queries x 64 keys)
    const h16 pa0 = ldA_f16(pP + (size_t)ln * 64, 0, g);
    const h16 pa1 = ldA_f16(pP + (size_t)ln * 64, 32, g);
    // V fragments from transposed store: lane row = output dim, keys contiguous
    const _Float16* vrow0 = vbase + (size_t)ln * NK + kb;
    const _Float16* vrow1 = vbase + (size_t)(16 + ln) * NK + kb;
    acc0 = WMMA_F16(pa0, *(const h16*)(vrow0 + 16 * g),      acc0);
    acc0 = WMMA_F16(pa1, *(const h16*)(vrow0 + 32 + 16 * g), acc0);
    acc1 = WMMA_F16(pa0, *(const h16*)(vrow1 + 16 * g),      acc1);
    acc1 = WMMA_F16(pa1, *(const h16*)(vrow1 + 32 + 16 * g), acc1);
    __syncthreads();
  }

  const int cbase = h * 32;
#pragma unroll
  for (int v = 0; v < 8; ++v) {
    const int pos = pos0 + 8 * g + v;
    if (pos < NQ) {
      const float inv = 1.0f / run_l[v];
      const size_t t = (size_t)pos * BB + b;
      att[t * CC + cbase + ln]      = (_Float16)(acc0[v] * inv);
      att[t * CC + cbase + 16 + ln] = (_Float16)(acc1[v] * inv);
    }
  }
}

// ---- kernel 6: out = identity + att@Wo^T + bo  (f32 output)
__global__ __launch_bounds__(32) void oproj_kernel(
    const _Float16* __restrict__ att, const _Float16* __restrict__ Wo,
    const float* __restrict__ bo, const float* __restrict__ identity,
    float* __restrict__ out) {
  const int lane = threadIdx.x, g = lane >> 4, ln = lane & 15;
  const int ct4 = blockIdx.x & 3;
  const int mt  = blockIdx.x >> 2;           // 0..224 (3600/16)
  const int t0 = mt * 16;
  const _Float16* arow = att + (size_t)(t0 + ln) * CC;
  const int c0 = ct4 * 64;
  const _Float16* brow[4];
#pragma unroll
  for (int j = 0; j < 4; ++j) brow[j] = Wo + (size_t)(c0 + 16 * j + ln) * CC;
  f8 acc[4] = {};
#pragma unroll
  for (int k0 = 0; k0 < CC; k0 += 32) {
    const h16 a = ldA_f16(arow, k0, g);
#pragma unroll
    for (int j = 0; j < 4; ++j)
      acc[j] = WMMA_F16(a, ldB_f16(brow[j], k0, g), acc[j]);
  }
#pragma unroll
  for (int j = 0; j < 4; ++j) {
    const int c = c0 + 16 * j + ln;
    const float bias = bo[c];
#pragma unroll
    for (int v = 0; v < 8; ++v) {
      const size_t t = (size_t)(t0 + 8 * g + v);
      out[t * CC + c] = acc[j][v] + bias + identity[t * CC + c];
    }
  }
}

extern "C" void kernel_launch(void* const* d_in, const int* in_sizes, int n_in,
                              void* d_out, int out_size, void* d_ws, size_t ws_size,
                              hipStream_t stream) {
  (void)in_sizes; (void)n_in; (void)out_size; (void)ws_size;
  const float* query = (const float*)d_in[0];
  const float* key   = (const float*)d_in[1];
  const float* value = (const float*)d_in[2];
  const float* qpos  = (const float*)d_in[3];
  const float* kpos  = (const float*)d_in[4];
  const float* qsine = (const float*)d_in[5];
  const float* Wqc = (const float*)d_in[6];   const float* bqc = (const float*)d_in[7];
  const float* Wqp = (const float*)d_in[8];   const float* bqp = (const float*)d_in[9];
  const float* Wqs = (const float*)d_in[10];  const float* bqs = (const float*)d_in[11];
  const float* Wkc = (const float*)d_in[12];  const float* bkc = (const float*)d_in[13];
  const float* Wkp = (const float*)d_in[14];  const float* bkp = (const float*)d_in[15];
  const float* Wv  = (const float*)d_in[16];  const float* bv  = (const float*)d_in[17];
  const float* Wo  = (const float*)d_in[18];  const float* bo  = (const float*)d_in[19];
  float* out = (float*)d_out;

  // f16 workspace: qh[B,H,NQ,64] kh[B,H,NK,64] vT[B,H,32,NK] att[3600,256] W[7][256*256]
  _Float16* ws  = (_Float16*)d_ws;
  _Float16* qh  = ws;                                   // 1,843,200 halves
  _Float16* kh  = qh + (size_t)BB * HH * NQ * 64;       // 8,388,608
  _Float16* vT  = kh + (size_t)BB * HH * NK * 64;       // 4,194,304
  _Float16* att = vT + (size_t)BB * HH * DD * NK;       // 921,600
  _Float16* hW  = att + (size_t)3600 * CC;              // 7*65,536  (~31.6 MB total)
  _Float16* hWqc = hW + 0 * 65536;
  _Float16* hWqp = hW + 1 * 65536;
  _Float16* hWqs = hW + 2 * 65536;
  _Float16* hWkc = hW + 3 * 65536;
  _Float16* hWkp = hW + 4 * 65536;
  _Float16* hWv  = hW + 5 * 65536;
  _Float16* hWo  = hW + 6 * 65536;

  const int ng = 65536 / 256;
  cvt_f16_kernel<<<ng, 256, 0, stream>>>(Wqc, hWqc, 65536);
  cvt_f16_kernel<<<ng, 256, 0, stream>>>(Wqp, hWqp, 65536);
  cvt_f16_kernel<<<ng, 256, 0, stream>>>(Wqs, hWqs, 65536);
  cvt_f16_kernel<<<ng, 256, 0, stream>>>(Wkc, hWkc, 65536);
  cvt_f16_kernel<<<ng, 256, 0, stream>>>(Wkp, hWkp, 65536);
  cvt_f16_kernel<<<ng, 256, 0, stream>>>(Wv,  hWv,  65536);
  cvt_f16_kernel<<<ng, 256, 0, stream>>>(Wo,  hWo,  65536);

  const int grid_q = 57 * BB * 4;      // 912 blocks, 16x64 strip each
  const int grid_k = 256 * BB * 4;     // 4096 blocks
  proj_q_kernel<<<grid_q, 32, 0, stream>>>(query, qpos, hWqc, hWqp, bqc, bqp, qh);
  proj_qs_kernel<<<grid_q, 32, 0, stream>>>(qsine, hWqs, bqs, qh);
  proj_k_kernel<<<grid_k, 32, 0, stream>>>(key, kpos, hWkc, hWkp, bkc, bkp, kh);
  proj_v_kernel<<<grid_k, 32, 0, stream>>>(value, hWv, bv, vT);
  attn_kernel<<<BB * HH * 57, 32, 0, stream>>>(qh, kh, vT, att);
  oproj_kernel<<<225 * 4, 32, 0, stream>>>(att, hWo, bo, query, out);
}
